// GCN_15401752724091
// MI455X (gfx1250) — matrix operand order repaired and checked by reading, not compiled, not run on hardware
//
#include <hip/hip_runtime.h>
#include <hip/hip_bf16.h>

// ---------------------------------------------------------------------------
// GCN forward for MI455X (gfx1250, wave32, WMMA, async-to-LDS staging).
//   conv GEMMs fuse the self-loop epilogue: agg = (h@W)*dinv^2, hwb = bf16(h@W)
//   scatter: agg[dst] += hwb[src]*norm  (fp32 atomics, L2-resident)
// ---------------------------------------------------------------------------

typedef __attribute__((ext_vector_type(16))) __bf16         v16bf;
typedef __attribute__((ext_vector_type(8)))  float          v8f;
typedef __attribute__((ext_vector_type(8)))  unsigned short u16x8;
typedef __attribute__((ext_vector_type(4)))  int            v4i;

#define AS1 __attribute__((address_space(1)))
#define AS3 __attribute__((address_space(3)))

#if __has_builtin(__builtin_amdgcn_global_load_async_to_lds_b128) && \
    __has_builtin(__builtin_amdgcn_s_wait_asynccnt)
#define HAVE_ASYNC 1
#else
#define HAVE_ASYNC 0
#endif

union BFrag { v16bf v; u16x8 h[2]; };

__device__ __forceinline__ unsigned short f2bf_bits(float f) {
  unsigned u = __float_as_uint(f);
  u += 0x7FFFu + ((u >> 16) & 1u);     // round-to-nearest-even to bf16
  return (unsigned short)(u >> 16);
}

// Build fragments from LDS (A, bf16 16x32 row-major) + global Wt, one WMMA.
// A frag: lane=row m=ln16; elems 0..7 -> K=hh*8+0..7; elems 8..15 -> K=16+hh*8+0..7.
// B frag: lane=col n=ln16; elems 0..7 -> K=hh*8+0..7; elems 8..15 -> K=16+hh*8+0..7.
__device__ __forceinline__ void mma_step(const unsigned short* lds_a,
                                         const unsigned short* __restrict__ Wt,
                                         int Kpad, int kbase, int nbase,
                                         int hh, int ln16, v8f& acc)
{
  __syncthreads();
  BFrag a, b;
  const unsigned short* ap = lds_a + (ln16 << 5) + (hh << 3);
  a.h[0] = *(const u16x8*)(ap);
  a.h[1] = *(const u16x8*)(ap + 16);
  const unsigned short* bp = Wt + (size_t)(nbase + ln16) * Kpad + kbase + (hh << 3);
  b.h[0] = *(const u16x8*)(bp);
  b.h[1] = *(const u16x8*)(bp + 16);
  acc = __builtin_amdgcn_wmma_f32_16x16x32_bf16(false, a.v, false, b.v,
                                                (short)0, acc, false, false);
  __syncthreads();
}

// Fused conv epilogue: agg = v*dinv^2 (self-loop init), hwb = bf16(v).
__device__ __forceinline__ void conv_epilogue(const v8f& acc, const float* __restrict__ dinv,
                                              float* __restrict__ agg,
                                              unsigned short* __restrict__ hwb,
                                              int mbase, int hh, int n)
{
  #pragma unroll
  for (int r = 0; r < 8; ++r) {
    const int m = mbase + (hh << 3) + r;
    const float v  = acc[r];
    const float dv = dinv[m];
    agg[(size_t)m * 128 + n] = v * dv * dv;
    hwb[(size_t)m * 128 + n] = f2bf_bits(v);
  }
}

// ---------------------------------------------------------------------------
// GEMM, fp32 A (conv1 only): blockDim = 256 (8 waves), 16 rows x 128 cols/block.
// ---------------------------------------------------------------------------
__global__ void k_gemm_f32A(const float* __restrict__ A,
                            const unsigned short* __restrict__ Wt,
                            int K, int Kpad,
                            const float* __restrict__ dinv,
                            float* __restrict__ agg,
                            unsigned short* __restrict__ hwb)
{
  __shared__ unsigned short lds_a[16 * 32];
  const int tid   = threadIdx.x;
  const int wave  = tid >> 5;
  const int lane  = tid & 31;
  const int hh    = lane >> 4;
  const int ln16  = lane & 15;
  const int mbase = blockIdx.x << 4;
  const int nbase = wave << 4;

  // this thread's fixed staging slot: row r, element pair c (blockDim == 256)
  const int sr = tid >> 4;
  const int sc = (tid & 15) << 1;
  const float* srow = A + (size_t)(mbase + sr) * K;

  v8f acc = {};

  const int full = K >> 5;                       // guard-free chunks
  for (int ck = 0; ck < full; ++ck) {
    const int k = (ck << 5) + sc;
    const float v0 = srow[k];
    const float v1 = srow[k + 1];
    ((unsigned*)lds_a)[tid] =
        (unsigned)f2bf_bits(v0) | ((unsigned)f2bf_bits(v1) << 16);
    if (k + 32 < K)
      __builtin_prefetch(srow + k + 32, 0, 1);
    mma_step(lds_a, Wt, Kpad, ck << 5, nbase, hh, ln16, acc);
  }
  if ((full << 5) < Kpad) {                      // guarded tail chunk
    const int kbase = full << 5;
    const int k = kbase + sc;
    const float v0 = (k     < K) ? srow[k]     : 0.0f;
    const float v1 = (k + 1 < K) ? srow[k + 1] : 0.0f;
    ((unsigned*)lds_a)[tid] =
        (unsigned)f2bf_bits(v0) | ((unsigned)f2bf_bits(v1) << 16);
    mma_step(lds_a, Wt, Kpad, kbase, nbase, hh, ln16, acc);
  }

  conv_epilogue(acc, dinv, agg, hwb, mbase, hh, nbase + ln16);
}

// ---------------------------------------------------------------------------
// GEMM, bf16 A (conv2 + FC head): A[M,Kpad] bf16 row-major, staged to LDS with
// GLOBAL_LOAD_ASYNC_TO_LDS_B128. blockDim = 32 * ceil(Nvalid/16).
// dinv != null -> fused conv epilogue (agg/hwb); else bias/relu store to Cout.
// ---------------------------------------------------------------------------
__global__ void k_gemm_bf16A(const unsigned short* __restrict__ A,
                             const unsigned short* __restrict__ Wt,
                             int Kpad, int Nvalid,
                             const float* __restrict__ bias, int relu, int out_bf16,
                             void* __restrict__ Cout,
                             const float* __restrict__ dinv,
                             float* __restrict__ agg,
                             unsigned short* __restrict__ hwb)
{
  __shared__ unsigned short lds_a[16 * 32];      // 1 KB tile
  const int tid   = threadIdx.x;
  const int wave  = tid >> 5;
  const int lane  = tid & 31;
  const int hh    = lane >> 4;
  const int ln16  = lane & 15;
  const int mbase = blockIdx.x << 4;
  const int nbase = wave << 4;

  v8f acc = {};

  const int nchunks = Kpad >> 5;
  for (int ck = 0; ck < nchunks; ++ck) {
    const int kbase = ck << 5;
#if HAVE_ASYNC
    if (tid < 64) {                              // 64 lanes x 16B = 1 KB tile DMA
      const int r  = tid >> 2;
      const int cg = (tid & 3) << 3;
      const unsigned short* gp = A + (size_t)(mbase + r) * Kpad + kbase + cg;
      __builtin_amdgcn_global_load_async_to_lds_b128(
          (AS1 v4i*)gp, (AS3 v4i*)(lds_a + (r << 5) + cg), 0, 0);
    }
    __builtin_amdgcn_s_wait_asynccnt(0);
#else
    for (int idx = tid; idx < 256; idx += blockDim.x) {
      const int r  = idx >> 4;
      const int c2 = (idx & 15) << 1;
      ((unsigned*)lds_a)[idx] =
          *(const unsigned*)(A + (size_t)(mbase + r) * Kpad + kbase + c2);
    }
#endif
    mma_step(lds_a, Wt, Kpad, kbase, nbase, hh, ln16, acc);
  }

  const int n = nbase + ln16;
  if (dinv) {
    conv_epilogue(acc, dinv, agg, hwb, mbase, hh, n);
  } else if (n < Nvalid) {
    const float bb = bias ? bias[n] : 0.0f;
    #pragma unroll
    for (int r = 0; r < 8; ++r) {
      const int m = mbase + (hh << 3) + r;
      float v = acc[r] + bb;
      if (relu) v = fmaxf(v, 0.0f);
      if (out_bf16)
        ((unsigned short*)Cout)[(size_t)m * Nvalid + n] = f2bf_bits(v);
      else
        ((float*)Cout)[(size_t)m * Nvalid + n] = v;
    }
  }
}

// ---------------------------------------------------------------------------
// Weight prep: W[K,N] fp32 row-major -> Wt[Npad,Kpad] bf16, zero padded.
// ---------------------------------------------------------------------------
__global__ void k_wconv(const float* __restrict__ W, unsigned short* __restrict__ Wt,
                        int K, int N, int Kpad, int Npad)
{
  const int i = blockIdx.x * blockDim.x + threadIdx.x;
  if (i >= Npad * Kpad) return;
  const int n = i / Kpad;
  const int k = i - n * Kpad;
  const float v = (n < N && k < K) ? W[(size_t)k * N + n] : 0.0f;
  Wt[i] = f2bf_bits(v);
}

// ---------------------------------------------------------------------------
// Graph-normalization kernels
// ---------------------------------------------------------------------------
__global__ void k_fill1(float* __restrict__ deg, int N) {
  const int i = blockIdx.x * blockDim.x + threadIdx.x;
  if (i < N) deg[i] = 1.0f;                         // self loop
}

__global__ void k_edge_deg(const int* __restrict__ dst, float* __restrict__ deg, int E) {
  const int i = blockIdx.x * blockDim.x + threadIdx.x;
  if (i < E)
    __hip_atomic_fetch_add(&deg[dst[i]], 1.0f, __ATOMIC_RELAXED, __HIP_MEMORY_SCOPE_AGENT);
}

__global__ void k_dinv(const float* __restrict__ deg, float* __restrict__ dinv, int N) {
  const int i = blockIdx.x * blockDim.x + threadIdx.x;
  if (i < N) { const float d = deg[i]; dinv[i] = d > 0.0f ? rsqrtf(d) : 0.0f; }
}

__global__ void k_norm(const int* __restrict__ src, const int* __restrict__ dst,
                       const float* __restrict__ dinv, float* __restrict__ nrm, int E) {
  const int i = blockIdx.x * blockDim.x + threadIdx.x;
  if (i < E) nrm[i] = dinv[src[i]] * dinv[dst[i]];
}

// One wave32 per edge; 4 bf16 per lane (8 B) covers 128 channels.
__global__ void k_scatter(const unsigned short* __restrict__ hwb,
                          const int* __restrict__ src, const int* __restrict__ dst,
                          const float* __restrict__ nrm, float* __restrict__ agg, int E)
{
  const int gid  = blockIdx.x * blockDim.x + threadIdx.x;
  const int e    = gid >> 5;
  const int lane = threadIdx.x & 31;
  if (e >= E) return;
  const int   s = src[e];
  const int   d = dst[e];
  const float w = nrm[e];
  const uint2 q = ((const uint2*)(hwb + (size_t)s * 128))[lane];
  const float x0 = __uint_as_float(q.x << 16);
  const float x1 = __uint_as_float(q.x & 0xFFFF0000u);
  const float x2 = __uint_as_float(q.y << 16);
  const float x3 = __uint_as_float(q.y & 0xFFFF0000u);
  float* pd = agg + (size_t)d * 128 + (lane << 2);
  __hip_atomic_fetch_add(pd + 0, x0 * w, __ATOMIC_RELAXED, __HIP_MEMORY_SCOPE_AGENT);
  __hip_atomic_fetch_add(pd + 1, x1 * w, __ATOMIC_RELAXED, __HIP_MEMORY_SCOPE_AGENT);
  __hip_atomic_fetch_add(pd + 2, x2 * w, __ATOMIC_RELAXED, __HIP_MEMORY_SCOPE_AGENT);
  __hip_atomic_fetch_add(pd + 3, x3 * w, __ATOMIC_RELAXED, __HIP_MEMORY_SCOPE_AGENT);
}

// h (bf16) = round( relu(agg + b) )
__global__ void k_bias_relu_bf16(const float* __restrict__ agg, const float* __restrict__ b,
                                 unsigned short* __restrict__ h, int NH) {
  const int i = blockIdx.x * blockDim.x + threadIdx.x;
  if (i < NH) h[i] = f2bf_bits(fmaxf(agg[i] + b[i & 127], 0.0f));
}

// ---------------------------------------------------------------------------
static inline int cdiv(long long a, long long b) { return (int)((a + b - 1) / b); }

extern "C" void kernel_launch(void* const* d_in, const int* in_sizes, int n_in,
                              void* d_out, int out_size, void* d_ws, size_t ws_size,
                              hipStream_t stream) {
  const float* x  = (const float*)d_in[0];
  const int*   ei = (const int*)  d_in[1];
  const float* W1 = (const float*)d_in[2];
  const float* b1 = (const float*)d_in[3];
  const float* W2 = (const float*)d_in[4];
  const float* b2 = (const float*)d_in[5];
  const float* fW[6] = { (const float*)d_in[6],  (const float*)d_in[8],
                         (const float*)d_in[10], (const float*)d_in[12],
                         (const float*)d_in[14], (const float*)d_in[16] };
  const float* fb[6] = { (const float*)d_in[7],  (const float*)d_in[9],
                         (const float*)d_in[11], (const float*)d_in[13],
                         (const float*)d_in[15], (const float*)d_in[17] };

  const int IN_DIM = 2129, HID_ = 128;
  const int N = in_sizes[0] / IN_DIM;     // 50000 (multiple of 16)
  const int E = in_sizes[1] / 2;          // 1600000
  const int* srcp = ei;
  const int* dstp = ei + E;
  const int Kp1 = ((IN_DIM + 31) / 32) * 32;   // 2144

  // ---- workspace carve (256B-aligned slots) ----
  char* p = (char*)d_ws;
  auto carve = [&](size_t bytes) -> char* {
    char* r = p; p += (bytes + 255) & ~(size_t)255; return r;
  };
  float*          deg  = (float*)carve((size_t)N * 4);
  float*          dinv = (float*)carve((size_t)N * 4);
  float*          nrm  = (float*)carve((size_t)E * 4);
  float*          agg  = (float*)carve((size_t)N * HID_ * 4);          // fp32 accum
  unsigned short* hwb  = (unsigned short*)carve((size_t)N * HID_ * 2); // bf16 h@W
  unsigned short* hA   = (unsigned short*)carve((size_t)N * HID_ * 2); // bf16 act
  unsigned short* f64a = (unsigned short*)carve((size_t)N * 64 * 2);
  unsigned short* f64b = (unsigned short*)carve((size_t)N * 64 * 2);
  unsigned short* W1t  = (unsigned short*)carve((size_t)128 * Kp1 * 2);
  unsigned short* W2t  = (unsigned short*)carve((size_t)128 * 128 * 2);
  const int fK[6] = {128, 64, 64, 64, 64, 64};
  const int fN[6] = {64, 64, 64, 64, 64, 30};
  unsigned short* fWt[6];
  int fNp[6];
  for (int i = 0; i < 6; ++i) {
    fNp[i] = ((fN[i] + 15) / 16) * 16;
    fWt[i] = (unsigned short*)carve((size_t)fNp[i] * fK[i] * 2);
  }

  // ---- weight conversion / transpose to bf16 ----
  k_wconv<<<cdiv((long long)128 * Kp1, 256), 256, 0, stream>>>(W1, W1t, IN_DIM, 128, Kp1, 128);
  k_wconv<<<cdiv(128 * 128, 256), 256, 0, stream>>>(W2, W2t, 128, 128, 128, 128);
  for (int i = 0; i < 6; ++i)
    k_wconv<<<cdiv(fNp[i] * fK[i], 256), 256, 0, stream>>>(fW[i], fWt[i], fK[i], fN[i], fK[i], fNp[i]);

  // ---- symmetric normalization ----
  k_fill1   <<<cdiv(N, 256), 256, 0, stream>>>(deg, N);
  k_edge_deg<<<cdiv(E, 256), 256, 0, stream>>>(dstp, deg, E);
  k_dinv    <<<cdiv(N, 256), 256, 0, stream>>>(deg, dinv, N);
  k_norm    <<<cdiv(E, 256), 256, 0, stream>>>(srcp, dstp, dinv, nrm, E);

  const dim3 gM(N / 16);            // 3125 row tiles
  const int  NH = N * HID_;
  const int  scat_grid = cdiv((long long)E * 32, 256);

  // ---- conv1 (fp32 x, fused self-loop epilogue) ----
  k_gemm_f32A<<<gM, 256, 0, stream>>>(x, W1t, IN_DIM, Kp1, dinv, agg, hwb);
  k_scatter  <<<scat_grid, 256, 0, stream>>>(hwb, srcp, dstp, nrm, agg, E);
  k_bias_relu_bf16<<<cdiv(NH, 256), 256, 0, stream>>>(agg, b1, hA, NH);

  // ---- conv2 x9 (shared weights, bf16 activations) ----
  for (int it = 0; it < 9; ++it) {
    k_gemm_bf16A<<<gM, 256, 0, stream>>>(hA, W2t, 128, 128,
                                         nullptr, 0, 0, nullptr, dinv, agg, hwb);
    k_scatter  <<<scat_grid, 256, 0, stream>>>(hwb, srcp, dstp, nrm, agg, E);
    k_bias_relu_bf16<<<cdiv(NH, 256), 256, 0, stream>>>(agg, b2, hA, NH);
  }

  // ---- FC head (bf16 chain, fp32 final) ----
  k_gemm_bf16A<<<gM, 128, 0, stream>>>(hA,   fWt[0], 128, 64, fb[0], 1, 1, f64a,
                                       nullptr, nullptr, nullptr);
  k_gemm_bf16A<<<gM, 128, 0, stream>>>(f64a, fWt[1],  64, 64, fb[1], 1, 1, f64b,
                                       nullptr, nullptr, nullptr);
  k_gemm_bf16A<<<gM, 128, 0, stream>>>(f64b, fWt[2],  64, 64, fb[2], 1, 1, f64a,
                                       nullptr, nullptr, nullptr);
  k_gemm_bf16A<<<gM, 128, 0, stream>>>(f64a, fWt[3],  64, 64, fb[3], 1, 1, f64b,
                                       nullptr, nullptr, nullptr);
  k_gemm_bf16A<<<gM, 128, 0, stream>>>(f64b, fWt[4],  64, 64, fb[4], 1, 1, f64a,
                                       nullptr, nullptr, nullptr);
  k_gemm_bf16A<<<gM,  64, 0, stream>>>(f64a, fWt[5],  64, 30, fb[5], 0, 0, d_out,
                                       nullptr, nullptr, nullptr);
}
